// SimpleGATV2_9208409883071
// MI455X (gfx1250) — compile-verified, matrix operand order checked by reading, not run on hardware
//
#include <hip/hip_runtime.h>

// ---------------------------------------------------------------------------
// GATv2 (3 layers, H=4, D=32, feat=128) for MI455X / gfx1250, wave32.
// GEMMs: v_wmma_f32_16x16x32_bf16, one wave holds the full K=128 A-slab in
// VGPRs and sweeps all 16 output column tiles (64 WMMAs / wave).
// Edge softmax/aggregation: L2-resident gathers + global float atomics
// (fs+fd = 51MB << 192MB L2).
// ---------------------------------------------------------------------------

typedef unsigned int  u32;
typedef unsigned short u16;

typedef __bf16 v16bf __attribute__((ext_vector_type(16)));
typedef float  v8f   __attribute__((ext_vector_type(8)));
typedef float  v4f   __attribute__((ext_vector_type(4)));
typedef u32    v4u   __attribute__((ext_vector_type(4)));

struct BVec { v4u lo, hi; };   // 32 bytes == 16 bf16 (one WMMA A/B operand)

#define N_NODES 50000
#define E_EDGES 800000
#define CDIM    128           // in/out feature dim for every layer
#define HEADS   4
#define DHEAD   32
#define NEG_SLOPE 0.2f
#define LN_EPS    1e-5f

// ---------------- small device helpers ----------------

__device__ __forceinline__ u16 f2bf(float f) {
    union { float f; u32 u; } v; v.f = f;
    u32 r = v.u + 0x7FFFu + ((v.u >> 16) & 1u);   // round-to-nearest-even
    return (u16)(r >> 16);
}

// order-preserving float<->uint mapping for atomicMax on floats
__device__ __forceinline__ u32 f2key(float f) {
    union { float f; u32 u; } v; v.f = f;
    return (v.u & 0x80000000u) ? ~v.u : (v.u | 0x80000000u);
}
__device__ __forceinline__ float key2f(u32 k) {
    union { float f; u32 u; } v;
    v.u = (k & 0x80000000u) ? (k & 0x7FFFFFFFu) : ~k;
    return v.f;
}

__device__ __forceinline__ int edge_src(int e, const int* src) {
    return (e < E_EDGES) ? src[e] : (e - E_EDGES);   // self-loop tail
}
__device__ __forceinline__ int edge_dst(int e, const int* dst) {
    return (e < E_EDGES) ? dst[e] : (e - E_EDGES);
}

// ---------------- activations f32 -> bf16 ----------------

__global__ void k_cvt_bf16(const float* __restrict__ x, u16* __restrict__ y, int n) {
    int t = blockIdx.x * blockDim.x + threadIdx.x;
    if (t < n) y[t] = f2bf(x[t]);
}

// ---------------- pack [W_src | W_dst] into WMMA B-operand layout ----------
// Layout: packed[ct(16)][kk(4)][lane(32)][i(16)] bf16, where for lane L:
//   N = ct_local*16 + (L&15), kbase = kk*32 + (L>>4)*8,
//   element i -> K = kbase + (i&7) + (i>>3)*16   (matches 16-bit B layout)

__global__ void k_pack_w(const float* __restrict__ wsrc,
                         const float* __restrict__ wdst,
                         u16* __restrict__ packed) {
    int t = blockIdx.x * blockDim.x + threadIdx.x;      // 16*4*32 = 2048
    if (t >= 16 * 4 * 32) return;
    int lane = t & 31;
    int ct   = t >> 7;                                  // 0..15
    const float* w = (ct < 8) ? wsrc : wdst;
    int n     = (ct & 7) * 16 + (lane & 15);
    int kbase = ((t >> 5) & 3) * 32 + (lane >> 4) * 8;
    u16* p = packed + (size_t)t * 16;
    #pragma unroll
    for (int i = 0; i < 16; ++i) {
        int k = kbase + (i & 7) + ((i >> 3) << 4);
        p[i] = f2bf(w[(size_t)k * CDIM + n]);
    }
}

// ---------------- WMMA GEMM: fs = Hbf16 @ Wsrc, fd = Hbf16 @ Wdst ----------
// One wave owns a 16-row M slab: the whole K=128 A operand lives in 32 VGPRs,
// then it sweeps the 16 output column tiles (8 fs + 8 fd) -> 64 WMMAs/wave.
// N_NODES = 50000 = 3125*16 -> no ragged tiles, EXEC stays all-ones.

__global__ void k_gemm_wmma(const u16* __restrict__ hb,
                            const u16* __restrict__ pw,
                            float* __restrict__ fs,
                            float* __restrict__ fd,
                            int nrows) {
    const int lane = threadIdx.x & 31;
    const int mt   = blockIdx.x * (blockDim.x >> 5) + (threadIdx.x >> 5);
    if (mt * 16 >= nrows) return;        // wave-uniform guard

    const int mrow  = mt * 16 + (lane & 15);
    const int klane = (lane >> 4) << 3;  // 0 or 8

    // Load full A slab: per lane, K = kk*32 + klane + {0..7, 16..23}
    v16bf a[4];
    #pragma unroll
    for (int kk = 0; kk < 4; ++kk) {
        const u16* ap = hb + (size_t)mrow * CDIM + kk * 32 + klane;
        BVec av;
        av.lo = *(const v4u*)(ap);        // 16B: K+0..7
        av.hi = *(const v4u*)(ap + 16);   // 16B: K+16..23
        a[kk] = __builtin_bit_cast(v16bf, av);
    }

    const int mo   = mt * 16 + ((lane >> 4) << 3);  // lanes 16-31 hold M=8..15
    const int ncol = lane & 15;

    #pragma unroll
    for (int ct = 0; ct < 16; ++ct) {    // 0..7 -> fs, 8..15 -> fd
        v8f c = {};
        #pragma unroll
        for (int kk = 0; kk < 4; ++kk) {
            const u16* bp = pw + (size_t)(((ct * 4 + kk) * 32 + lane) * 16);
            BVec bv;
            bv.lo = *(const v4u*)(bp);
            bv.hi = *(const v4u*)(bp + 8);
            v16bf b = __builtin_bit_cast(v16bf, bv);
            c = __builtin_amdgcn_wmma_f32_16x16x32_bf16(
                    /*neg_a=*/false, a[kk], /*neg_b=*/false, b,
                    /*c_mod=*/(short)0, c, /*reuse_a=*/false, /*reuse_b=*/false);
        }
        float* outp = (ct < 8) ? fs : fd;
        const int col = ((ct & 7) << 4) + ncol;
        #pragma unroll
        for (int r = 0; r < 8; ++r)
            outp[(size_t)(mo + r) * CDIM + col] = c[r];
    }
}

// ---------------- edge phase: logits + segment max ----------------
// thread t -> (edge e = t/4, head h = t&3); 32 d-values per thread via v4f.

__global__ void k_logits(const float* __restrict__ fs, const float* __restrict__ fd,
                         const int* __restrict__ src, const int* __restrict__ dst,
                         const float* __restrict__ attn,
                         float* __restrict__ elog, u32* __restrict__ segmax) {
    int t = blockIdx.x * blockDim.x + threadIdx.x;
    int e = t >> 2, h = t & 3;
    if (e >= E_EDGES + N_NODES) return;
    int s = edge_src(e, src);
    int d = edge_dst(e, dst);
    const v4f* a = (const v4f*)(fs + (size_t)s * CDIM + h * DHEAD);
    const v4f* b = (const v4f*)(fd + (size_t)d * CDIM + h * DHEAD);
    const v4f* w = (const v4f*)(attn + h * DHEAD);
    float acc = 0.f;
    #pragma unroll
    for (int i = 0; i < 8; ++i) {
        v4f m = a[i] + b[i];
        v4f wr = w[i];
        float m0 = m.x > 0.f ? m.x : NEG_SLOPE * m.x;
        float m1 = m.y > 0.f ? m.y : NEG_SLOPE * m.y;
        float m2 = m.z > 0.f ? m.z : NEG_SLOPE * m.z;
        float m3 = m.w > 0.f ? m.w : NEG_SLOPE * m.w;
        acc += m0 * wr.x + m1 * wr.y + m2 * wr.z + m3 * wr.w;
    }
    elog[(size_t)e * HEADS + h] = acc;
    atomicMax(&segmax[(size_t)d * HEADS + h], f2key(acc));
}

// ---------------- edge phase: exp + segment sum ----------------

__global__ void k_expnorm(const int* __restrict__ dst,
                          float* __restrict__ elog,
                          const u32* __restrict__ segmax,
                          float* __restrict__ denom) {
    int t = blockIdx.x * blockDim.x + threadIdx.x;
    int e = t >> 2, h = t & 3;
    if (e >= E_EDGES + N_NODES) return;
    int d = edge_dst(e, dst);
    float mx = key2f(segmax[(size_t)d * HEADS + h]);
    float ex = __expf(elog[(size_t)e * HEADS + h] - mx);
    elog[(size_t)e * HEADS + h] = ex;               // reuse buffer for exp()
    atomicAdd(&denom[(size_t)d * HEADS + h], ex);
}

// ---------------- edge phase: alpha-weighted scatter-add ----------------

__global__ void k_aggregate(const float* __restrict__ fs,
                            const int* __restrict__ src, const int* __restrict__ dst,
                            const float* __restrict__ elog,
                            const float* __restrict__ denom,
                            float* __restrict__ outb) {
    int t = blockIdx.x * blockDim.x + threadIdx.x;
    int e = t >> 2, h = t & 3;
    if (e >= E_EDGES + N_NODES) return;
    int s = edge_src(e, src);
    int d = edge_dst(e, dst);
    float alpha = elog[(size_t)e * HEADS + h] / denom[(size_t)d * HEADS + h];
    const v4f* a = (const v4f*)(fs + (size_t)s * CDIM + h * DHEAD);
    float* o = outb + (size_t)d * CDIM + h * DHEAD;
    #pragma unroll
    for (int i = 0; i < 8; ++i) {
        v4f v = a[i];
        atomicAdd(o + i * 4 + 0, alpha * v.x);
        atomicAdd(o + i * 4 + 1, alpha * v.y);
        atomicAdd(o + i * 4 + 2, alpha * v.z);
        atomicAdd(o + i * 4 + 3, alpha * v.w);
    }
}

// ---------------- bias + LayerNorm + ELU, emit next-layer bf16 ----------

__global__ void k_finalize(const float* __restrict__ outb,
                           const float* __restrict__ bias,
                           const float* __restrict__ gamma,
                           const float* __restrict__ beta,
                           u16* __restrict__ hb,
                           float* __restrict__ out_f32 /* null except last layer */) {
    __shared__ float s1[CDIM];
    __shared__ float s2[CDIM];
    int n = blockIdx.x;
    int c = threadIdx.x;              // 128 threads
    float x = outb[(size_t)n * CDIM + c] + bias[c];
    s1[c] = x;
    s2[c] = x * x;
    __syncthreads();
    #pragma unroll
    for (int off = CDIM / 2; off > 0; off >>= 1) {
        if (c < off) { s1[c] += s1[c + off]; s2[c] += s2[c + off]; }
        __syncthreads();
    }
    float mu  = s1[0] * (1.0f / CDIM);
    float var = s2[0] * (1.0f / CDIM) - mu * mu;
    float y = (x - mu) * rsqrtf(var + LN_EPS) * gamma[c] + beta[c];
    float z = y > 0.f ? y : (__expf(y) - 1.f);      // ELU
    hb[(size_t)n * CDIM + c] = f2bf(z);
    if (out_f32) out_f32[(size_t)n * CDIM + c] = z;
}

// ---------------------------------------------------------------------------

extern "C" void kernel_launch(void* const* d_in, const int* in_sizes, int n_in,
                              void* d_out, int out_size, void* d_ws, size_t ws_size,
                              hipStream_t stream) {
    (void)in_sizes; (void)n_in; (void)out_size; (void)ws_size;

    const float* features = (const float*)d_in[0];
    const int*   src      = (const int*)d_in[1];
    const int*   dst      = (const int*)d_in[2];

    const int EP  = E_EDGES + N_NODES;           // edges incl. self loops
    const int EP4 = EP * HEADS;

    // ---- workspace carve-out (256B aligned) ----
    char*  ws  = (char*)d_ws;
    size_t off = 0;
    auto carve = [&](size_t bytes) -> void* {
        off = (off + 255) & ~(size_t)255;
        void* p = ws + off;
        off += bytes;
        return p;
    };
    u16*   hb     = (u16*)  carve((size_t)N_NODES * CDIM * sizeof(u16));
    float* fs     = (float*)carve((size_t)N_NODES * CDIM * sizeof(float));
    float* fd     = (float*)carve((size_t)N_NODES * CDIM * sizeof(float));
    float* outb   = (float*)carve((size_t)N_NODES * CDIM * sizeof(float));
    float* elog   = (float*)carve((size_t)EP4 * sizeof(float));
    u32*   segmax = (u32*)  carve((size_t)N_NODES * HEADS * sizeof(u32));
    float* denom  = (float*)carve((size_t)N_NODES * HEADS * sizeof(float));
    u16*   pw[3];
    for (int l = 0; l < 3; ++l)
        pw[l] = (u16*)carve((size_t)16 * 4 * 32 * 16 * sizeof(u16));

    // ---- activations -> bf16; weights -> packed WMMA layout ----
    k_cvt_bf16<<<(N_NODES * CDIM + 255) / 256, 256, 0, stream>>>(
        features, hb, N_NODES * CDIM);
    for (int l = 0; l < 3; ++l) {
        const float* wsrc = (const float*)d_in[3 + 6 * l];
        const float* wdst = (const float*)d_in[4 + 6 * l];
        k_pack_w<<<8, 256, 0, stream>>>(wsrc, wdst, pw[l]);
    }

    const int m_waves     = N_NODES / 16;                 // 3125 M slabs
    const int gemm_blocks = (m_waves + 7) / 8;            // 8 waves / block
    const int edge_blocks = (EP4 + 255) / 256;

    for (int l = 0; l < 3; ++l) {
        const float* attn  = (const float*)d_in[5 + 6 * l];
        const float* bias  = (const float*)d_in[6 + 6 * l];
        const float* gamma = (const float*)d_in[7 + 6 * l];
        const float* beta  = (const float*)d_in[8 + 6 * l];

        hipMemsetAsync(segmax, 0, (size_t)N_NODES * HEADS * sizeof(u32),  stream);
        hipMemsetAsync(denom,  0, (size_t)N_NODES * HEADS * sizeof(float), stream);
        hipMemsetAsync(outb,   0, (size_t)N_NODES * CDIM  * sizeof(float), stream);

        k_gemm_wmma<<<gemm_blocks, 256, 0, stream>>>(hb, pw[l], fs, fd, N_NODES);
        k_logits<<<edge_blocks, 256, 0, stream>>>(fs, fd, src, dst, attn, elog, segmax);
        k_expnorm<<<edge_blocks, 256, 0, stream>>>(dst, elog, segmax, denom);
        k_aggregate<<<edge_blocks, 256, 0, stream>>>(fs, src, dst, elog, denom, outb);
        k_finalize<<<N_NODES, CDIM, 0, stream>>>(
            outb, bias, gamma, beta, hb,
            (l == 2) ? (float*)d_out : nullptr);
    }
}
